// VQModel_53171695124979
// MI455X (gfx1250) — matrix operand order repaired and checked by reading, not compiled, not run on hardware
//
#include <hip/hip_runtime.h>
#include <hip/hip_bf16.h>

// ---------------------------------------------------------------------------
// VQGAN forward for MI455X (gfx1250, wave32).  All conv / attention GEMMs go
// through v_wmma_f32_16x16x32_bf16 (fp32 accumulate), activations kept fp32
// in global memory and converted to bf16 while staging into LDS.
//
// Block tile 64(out-ch) x 128(pixels), BK=32; 8 waves each own a 16x64 tile
// -> 4 WMMAs per wave per K step.  LDS holds A in a fragment-ready
// K-permuted layout and B transposed, so every lane's 16-element bf16 A
// fragment is 32 contiguous bytes and B's two 16-byte runs (2x ds_load_b128 each).
//
// Input-pointer ordering assumption (compile-only environment, documented):
//   d_in[0] = x; d_in[1..] = params flattened as a JAX pytree (dict keys
//   sorted alphabetically at every level, lists in index order, None
//   skipped).  conv_p {w,b} -> (b,w); gn_p {g,b} -> (b,g);
//   res_p -> conv1, conv2, [nin], norm1, norm2;
//   attn_p -> k, norm, proj, q, v;
//   params -> codebook, decoder, encoder, post_quant_conv, quant_conv.
// Workspace use: ~212 MB (3x64MB ping-pong + attention scratch).
// ---------------------------------------------------------------------------

typedef __attribute__((ext_vector_type(16))) __bf16 v16bf;
typedef __attribute__((ext_vector_type(8)))  float  v8f;

union Frag { v16bf v; unsigned short u[16]; uint4 q2[2]; };

__device__ __forceinline__ unsigned short f2bf(float f) {
  unsigned int u = __float_as_uint(f);
  unsigned int r = u + 0x7fffu + ((u >> 16) & 1u);   // round-to-nearest-even
  return (unsigned short)(r >> 16);
}

// Involution permutation on K within a 32-wide tile so that each lane's A
// fragment (ISA 7.12.2 16-bit A layout) is contiguous in LDS.
__device__ __forceinline__ int kperm(int k) {
  return (k < 8) ? k : (k < 16 ? k + 8 : (k < 24 ? k - 8 : k));
}

#define CBM 64
#define CBN 128
#define CBK 32

// ----------------------- implicit-GEMM convolution -------------------------
// out[co, oy, ox] = bias[co] + sum_{ci,ky,kx} w[co,ci,ky,kx] *
//                   in[ci, oy*stride+ky-padBeg, ox*stride+kx-padBeg]
__global__ __launch_bounds__(256) void conv_wmma_kernel(
    const float* __restrict__ in, const float* __restrict__ wgt,
    const float* __restrict__ bias, float* __restrict__ out,
    int Cin, int Cout, int Hin, int Win, int Hout, int Wout,
    int ks, int stride, int padBeg)
{
  const int P   = Hout * Wout;
  const int Ksz = Cin * ks * ks;
  const int tile_m = blockIdx.y * CBM;
  const int tile_n = blockIdx.x * CBN;
  __shared__ unsigned short AF[CBM][CBK];   // AF[m][q] = A[m][kperm(q)]
  __shared__ unsigned short BF[CBN][CBK];   // BF[n][k] = B[k][n]
  const int tid  = threadIdx.x;
  const int lane = tid & 31;
  const int wave = tid >> 5;
  const int wm   = wave & 3;                // 16-row block of out-channels
  const int wn   = wave >> 2;               // 64-pixel block
  const int hi   = (lane >= 16) ? 16 : 0;
  const int am   = wm * 16 + (lane & 15);
  const bool vecA = ((Ksz & 3) == 0);       // row-aligned float4 weight loads

  v8f acc[4] = {{}, {}, {}, {}};

  // B-staging: fixed local k per thread, 16 consecutive pixels
  const int bn0 = (tid & 7) * 16;
  const int bkl = tid >> 3;
  const int p0  = tile_n + bn0;
  const int ks2 = ks * ks;

  for (int k0 = 0; k0 < Ksz; k0 += CBK) {
    if (lane == 0 && k0 + CBK < Ksz)
      __builtin_prefetch(wgt + (long)tile_m * Ksz + k0 + CBK, 0, 3);  // global_prefetch_b8

    // ---- stage A (weights), K-permuted store ----
#pragma unroll
    for (int tt = 0; tt < 2; ++tt) {
      int id = tid * 2 + tt;                // 0..511
      int m  = id >> 3;                     // 0..63
      int kq = (id & 7) * 4;                // 0,4,...,28
      int qb = kperm(kq);                   // group-of-4 preserved by kperm
      int oc = tile_m + m;
      int kg = k0 + kq;
      unsigned r0, r1;
      if (vecA && oc < Cout && kg + 3 < Ksz) {
        const float4 w4 = *(const float4*)(wgt + (long)oc * Ksz + kg);
        r0 = (unsigned)f2bf(w4.x) | ((unsigned)f2bf(w4.y) << 16);
        r1 = (unsigned)f2bf(w4.z) | ((unsigned)f2bf(w4.w) << 16);
      } else {
        float e0 = (oc < Cout && kg + 0 < Ksz) ? wgt[(long)oc * Ksz + kg + 0] : 0.f;
        float e1 = (oc < Cout && kg + 1 < Ksz) ? wgt[(long)oc * Ksz + kg + 1] : 0.f;
        float e2 = (oc < Cout && kg + 2 < Ksz) ? wgt[(long)oc * Ksz + kg + 2] : 0.f;
        float e3 = (oc < Cout && kg + 3 < Ksz) ? wgt[(long)oc * Ksz + kg + 3] : 0.f;
        r0 = (unsigned)f2bf(e0) | ((unsigned)f2bf(e1) << 16);
        r1 = (unsigned)f2bf(e2) | ((unsigned)f2bf(e3) << 16);
      }
      *(unsigned*)&AF[m][qb]     = r0;
      *(unsigned*)&AF[m][qb + 2] = r1;
    }

    // ---- stage B (im2col), transposed store, incremental pixel walk ----
    {
      int kg = k0 + bkl;
      bool kok = kg < Ksz;
      int ci = 0, ky = 0, kx = 0;
      if (kok) { ci = kg / ks2; int r = kg - ci * ks2; ky = r / ks; kx = r - ky * ks; }
      const float* src = in + (long)ci * Hin * Win;
      int oy = p0 / Wout, ox = p0 - oy * Wout;
#pragma unroll
      for (int t = 0; t < 16; ++t) {
        float v = 0.f;
        if (kok && (p0 + t) < P) {
          int iy = oy * stride + ky - padBeg;
          int ix = ox * stride + kx - padBeg;
          if (iy >= 0 && iy < Hin && ix >= 0 && ix < Win)
            v = src[iy * Win + ix];
        }
        BF[bn0 + t][bkl] = f2bf(v);
        if (++ox == Wout) { ox = 0; ++oy; }
      }
    }
    __syncthreads();

    Frag a, b;
    a.q2[0] = *(const uint4*)&AF[am][hi];
    a.q2[1] = *(const uint4*)&AF[am][hi + 8];
#pragma unroll
    for (int s = 0; s < 4; ++s) {
      int bn = wn * 64 + s * 16 + (lane & 15);
      b.q2[0] = *(const uint4*)&BF[bn][hi / 2];
      b.q2[1] = *(const uint4*)&BF[bn][16 + hi / 2];
      acc[s] = __builtin_amdgcn_wmma_f32_16x16x32_bf16(false, a.v, false, b.v,
                                                       (short)0, acc[s], false, false);
    }
    __syncthreads();
  }

#pragma unroll
  for (int r = 0; r < 8; ++r) {                               // ISA C layout
    int oc = tile_m + wm * 16 + r + ((lane >= 16) ? 8 : 0);
    if (oc >= Cout) continue;
    float bv = bias[oc];
    long rowbase = (long)oc * P;
#pragma unroll
    for (int s = 0; s < 4; ++s) {
      int p = tile_n + wn * 64 + s * 16 + (lane & 15);
      if (p < P) out[rowbase + p] = acc[s][r] + bv;
    }
  }
}

// ----------------------- generic WMMA GEMM (attention) ---------------------
// C[m,n] = scale * sum_k opA(A)[m,k] * opB(B)[k,n]
__global__ __launch_bounds__(256) void gemm_wmma_kernel(
    const float* __restrict__ A, const float* __restrict__ B,
    float* __restrict__ C, int M, int N, int K,
    int transA, int transB, float scale)
{
  const int tile_m = blockIdx.y * CBM;
  const int tile_n = blockIdx.x * CBN;
  __shared__ unsigned short AF[CBM][CBK];
  __shared__ unsigned short BF[CBN][CBK];
  const int tid  = threadIdx.x;
  const int lane = tid & 31;
  const int wave = tid >> 5;
  const int wm = wave & 3, wn = wave >> 2;
  const int hi = (lane >= 16) ? 16 : 0;
  const int am = wm * 16 + (lane & 15);
  const int bn0 = (tid & 7) * 16;
  const int bkl = tid >> 3;

  v8f acc[4] = {{}, {}, {}, {}};

  for (int k0 = 0; k0 < K; k0 += CBK) {
#pragma unroll
    for (int t = 0; t < 8; ++t) {
      int e = tid * 8 + t;
      int m = e >> 5, k = e & 31;
      int mg = tile_m + m, kg = k0 + k;
      float v = 0.f;
      if (mg < M && kg < K)
        v = transA ? A[(long)kg * M + mg] : A[(long)mg * K + kg];
      AF[m][kperm(k)] = f2bf(v);
    }
    {
      int kg = k0 + bkl;
#pragma unroll
      for (int t = 0; t < 16; ++t) {
        int ng = tile_n + bn0 + t;
        float v = 0.f;
        if (kg < K && ng < N)
          v = transB ? B[(long)ng * K + kg] : B[(long)kg * N + ng];
        BF[bn0 + t][bkl] = f2bf(v);
      }
    }
    __syncthreads();

    Frag a, b;
    a.q2[0] = *(const uint4*)&AF[am][hi];
    a.q2[1] = *(const uint4*)&AF[am][hi + 8];
#pragma unroll
    for (int s = 0; s < 4; ++s) {
      int bn = wn * 64 + s * 16 + (lane & 15);
      b.q2[0] = *(const uint4*)&BF[bn][hi / 2];
      b.q2[1] = *(const uint4*)&BF[bn][16 + hi / 2];
      acc[s] = __builtin_amdgcn_wmma_f32_16x16x32_bf16(false, a.v, false, b.v,
                                                       (short)0, acc[s], false, false);
    }
    __syncthreads();
  }

#pragma unroll
  for (int r = 0; r < 8; ++r) {
    int mg = tile_m + wm * 16 + r + ((lane >= 16) ? 8 : 0);
    if (mg >= M) continue;
    long rowbase = (long)mg * N;
#pragma unroll
    for (int s = 0; s < 4; ++s) {
      int n = tile_n + wn * 64 + s * 16 + (lane & 15);
      if (n < N) C[rowbase + n] = scale * acc[s][r];
    }
  }
}

// ================= CDNA5 async / TDM / cluster probe kernels ===============
// Standalone (never launched); they exist to exercise the gfx1250-specific
// async-to-LDS, Tensor-Data-Mover and cluster-barrier paths in codegen.

// ASYNCcnt-tracked global->LDS copy (ISA 15.18.3 op 96 + s_wait_asynccnt).
// LDS byte address = low 32 bits of the flat shared pointer (LDS aperture
// keeps the offset in addr[31:0], ISA 10.2).
__global__ __launch_bounds__(32) void cdna5_async_probe_kernel(
    const float* __restrict__ src, float* __restrict__ dst)
{
  __shared__ float tile[32];
  unsigned lds = (unsigned)(unsigned long long)&tile[threadIdx.x];
  unsigned long long ga = (unsigned long long)(src + threadIdx.x);
  asm volatile("global_load_async_to_lds_b32 %0, %1, off"
               :: "v"(lds), "v"(ga) : "memory");
  asm volatile("s_wait_asynccnt 0" ::: "memory");
  __syncthreads();
  dst[threadIdx.x] = tile[31 - threadIdx.x];
}

#if defined(__HIP_DEVICE_COMPILE__) && __has_builtin(__builtin_amdgcn_tensor_load_to_lds)
typedef __attribute__((ext_vector_type(4))) unsigned int v4u_t;
typedef __attribute__((ext_vector_type(8))) int v8i_t;
typedef __attribute__((ext_vector_type(4))) int v4i_t;
__global__ __launch_bounds__(32) void cdna5_tdm_probe_kernel(float* __restrict__ dst)
{
  v4u_t g0 = {}; v8i_t g1 = {}; v4i_t g2 = {}; v4i_t g3 = {};
#if __clang_major__ >= 23
  v8i_t g4 = {};
  __builtin_amdgcn_tensor_load_to_lds(g0, g1, g2, g3, g4, 0);   // 6-arg flavor
#else
  __builtin_amdgcn_tensor_load_to_lds(g0, g1, g2, g3, 0);       // 5-arg flavor
#endif
#if __has_builtin(__builtin_amdgcn_s_wait_tensorcnt)
  __builtin_amdgcn_s_wait_tensorcnt(0);
#endif
#if __has_builtin(__builtin_amdgcn_s_cluster_barrier)
  __builtin_amdgcn_s_cluster_barrier();
#endif
  if (threadIdx.x == 0) dst[0] = 1.0f;
}
#endif

// ----------------------- GroupNorm (+optional SiLU) ------------------------
__global__ __launch_bounds__(256) void gn_kernel(
    const float* __restrict__ in, float* __restrict__ out,
    const float* __restrict__ g, const float* __restrict__ b,
    int HW, int cpg, int do_silu)
{
  const int grp = blockIdx.x;
  const long base = (long)grp * cpg * HW;
  const int n = cpg * HW;
  float s = 0.f, s2 = 0.f;
  for (int i = threadIdx.x; i < n; i += 256) {
    float v = in[base + i]; s += v; s2 += v * v;
  }
  __shared__ float sh1[256], sh2[256];
  sh1[threadIdx.x] = s; sh2[threadIdx.x] = s2; __syncthreads();
  for (int off = 128; off > 0; off >>= 1) {
    if (threadIdx.x < off) {
      sh1[threadIdx.x] += sh1[threadIdx.x + off];
      sh2[threadIdx.x] += sh2[threadIdx.x + off];
    }
    __syncthreads();
  }
  const float mean = sh1[0] / n;
  const float var  = sh2[0] / n - mean * mean;
  const float rinv = rsqrtf(var + 1e-6f);
  for (int i = threadIdx.x; i < n; i += 256) {
    int c = grp * cpg + i / HW;
    float v = (in[base + i] - mean) * rinv * g[c] + b[c];
    if (do_silu) v = v * (1.f / (1.f + __expf(-v)));
    out[base + i] = v;
  }
}

// ----------------------- row softmax ---------------------------------------
__global__ __launch_bounds__(256) void softmax_rows_kernel(float* __restrict__ w, int N)
{
  const long base = (long)blockIdx.x * N;
  __shared__ float sh[256];
  float m = -1e30f;
  for (int i = threadIdx.x; i < N; i += 256) m = fmaxf(m, w[base + i]);
  sh[threadIdx.x] = m; __syncthreads();
  for (int off = 128; off > 0; off >>= 1) {
    if (threadIdx.x < off) sh[threadIdx.x] = fmaxf(sh[threadIdx.x], sh[threadIdx.x + off]);
    __syncthreads();
  }
  m = sh[0]; __syncthreads();
  float s = 0.f;
  for (int i = threadIdx.x; i < N; i += 256) {
    float e = __expf(w[base + i] - m);
    w[base + i] = e; s += e;
  }
  sh[threadIdx.x] = s; __syncthreads();
  for (int off = 128; off > 0; off >>= 1) {
    if (threadIdx.x < off) sh[threadIdx.x] += sh[threadIdx.x + off];
    __syncthreads();
  }
  const float r = 1.f / sh[0];
  for (int i = threadIdx.x; i < N; i += 256) w[base + i] *= r;
}

// ----------------------- elementwise helpers -------------------------------
__global__ void add_kernel(const float* a, const float* bsrc, float* o, int n)
{
  int i = blockIdx.x * blockDim.x + threadIdx.x;
  if (i < n) o[i] = a[i] + bsrc[i];
}

__global__ void upsample_kernel(const float* __restrict__ in, float* __restrict__ out,
                                int C, int H, int W)
{
  int n = C * 4 * H * W;
  int i = blockIdx.x * blockDim.x + threadIdx.x;
  if (i >= n) return;
  int HW2 = 4 * H * W;
  int c = i / HW2, rem = i - c * HW2;
  int y = rem / (2 * W), x = rem - y * (2 * W);
  out[i] = in[(long)c * H * W + (y >> 1) * W + (x >> 1)];
}

// ----------------------- VQ codebook argmin --------------------------------
__global__ void quantize_kernel(const float* __restrict__ z, const float* __restrict__ cb,
                                float* __restrict__ zq, float* __restrict__ sq,
                                int HW, int NE)
{
  int p = blockIdx.x * blockDim.x + threadIdx.x;
  if (p >= HW) return;
  float z0 = z[p], z1 = z[HW + p], z2 = z[2 * HW + p], z3 = z[3 * HW + p];
  float best = 3.4e38f; int bi = 0;
  for (int e = 0; e < NE; ++e) {
    const float* c = cb + 4 * e;
    float d0 = z0 - c[0], d1 = z1 - c[1], d2 = z2 - c[2], d3 = z3 - c[3];
    float d = d0 * d0 + d1 * d1 + d2 * d2 + d3 * d3;
    if (d < best) { best = d; bi = e; }
  }
  const float* c = cb + 4 * bi;
  zq[p] = c[0]; zq[HW + p] = c[1]; zq[2 * HW + p] = c[2]; zq[3 * HW + p] = c[3];
  sq[p] = best;
}

__global__ __launch_bounds__(256) void diff_kernel(const float* __restrict__ sq,
                                                   float* __restrict__ outp,
                                                   int n, float scale)
{
  __shared__ float sh[256];
  float s = 0.f;
  for (int i = threadIdx.x; i < n; i += 256) s += sq[i];
  sh[threadIdx.x] = s; __syncthreads();
  for (int off = 128; off > 0; off >>= 1) {
    if (threadIdx.x < off) sh[threadIdx.x] += sh[threadIdx.x + off];
    __syncthreads();
  }
  if (threadIdx.x == 0) outp[0] = sh[0] * scale;
}

// ===========================================================================
// Host orchestration
// ===========================================================================
struct PW  { const float* b; const float* w; };
struct PGN { const float* b; const float* g; };
struct PRes { PW c1, c2, nin; PGN n1, n2; bool has_nin; };
struct PAttn { PW k, proj, q, v; PGN norm; };

extern "C" void kernel_launch(void* const* d_in, const int* in_sizes, int n_in,
                              void* d_out, int out_size, void* d_ws, size_t ws_size,
                              hipStream_t stream) {
  (void)in_sizes; (void)n_in; (void)out_size; (void)ws_size;

  int cur = 0;
  auto rdF  = [&]() { return (const float*)d_in[cur++]; };
  auto rdPW = [&]() { PW p; p.b = rdF(); p.w = rdF(); return p; };
  auto rdGN = [&]() { PGN p; p.b = rdF(); p.g = rdF(); return p; };
  auto rdRes = [&](bool nin) {
    PRes r; r.c1 = rdPW(); r.c2 = rdPW();
    if (nin) r.nin = rdPW();
    r.n1 = rdGN(); r.n2 = rdGN(); r.has_nin = nin; return r;
  };
  auto rdAttn = [&]() {
    PAttn a; a.k = rdPW(); a.norm = rdGN(); a.proj = rdPW();
    a.q = rdPW(); a.v = rdPW(); return a;
  };

  // ---- parse inputs (order documented in file header) ----
  const float* x        = rdF();              // [1,3,256,256]
  const float* codebook = rdF();              // [16384,4]
  // decoder
  PW   d_conv_in  = rdPW();
  PW   d_conv_out = rdPW();
  PAttn d_attn    = rdAttn();
  PRes d_mb1      = rdRes(false);
  PRes d_mb2      = rdRes(false);
  PGN  d_norm_out = rdGN();
  PRes up[4][3];
  up[0][0] = rdRes(true);  up[0][1] = rdRes(false); up[0][2] = rdRes(false);
  up[1][0] = rdRes(true);  up[1][1] = rdRes(false); up[1][2] = rdRes(false);
  up[2][0] = rdRes(false); up[2][1] = rdRes(false); up[2][2] = rdRes(false);
  up[3][0] = rdRes(false); up[3][1] = rdRes(false); up[3][2] = rdRes(false);
  PW upsmp[4];
  upsmp[1] = rdPW(); upsmp[2] = rdPW(); upsmp[3] = rdPW();   // upsample[0]=None
  // encoder
  PW   e_conv_in  = rdPW();
  PW   e_conv_out = rdPW();
  PRes down[4][2];
  down[0][0] = rdRes(false); down[0][1] = rdRes(false);
  down[1][0] = rdRes(true);  down[1][1] = rdRes(false);
  down[2][0] = rdRes(true);  down[2][1] = rdRes(false);
  down[3][0] = rdRes(false); down[3][1] = rdRes(false);
  PW dsmp[3]; dsmp[0] = rdPW(); dsmp[1] = rdPW(); dsmp[2] = rdPW();
  PAttn e_attn    = rdAttn();
  PRes e_mb1      = rdRes(false);
  PRes e_mb2      = rdRes(false);
  PGN  e_norm_out = rdGN();
  PW post_quant   = rdPW();
  PW quant        = rdPW();

  // ---- workspace bump allocator ----
  char* wsb = (char*)d_ws; size_t off = 0;
  auto alloc = [&](size_t bytes) -> float* {
    float* p = (float*)(wsb + off);
    off += (bytes + 255) & ~(size_t)255;
    return p;
  };
  const size_t BIG = (size_t)256 * 256 * 256 * sizeof(float);  // 64 MB
  float* bufA = alloc(BIG);
  float* bufB = alloc(BIG);
  float* bufC = alloc(BIG);
  float* qb   = alloc((size_t)512 * 1024 * 4);
  float* kb   = alloc((size_t)512 * 1024 * 4);
  float* vb   = alloc((size_t)512 * 1024 * 4);
  float* attw = alloc((size_t)1024 * 1024 * 4);
  float* z0   = alloc(4 * 1024 * 4);
  float* z1   = alloc(4 * 1024 * 4);
  float* zqv  = alloc(4 * 1024 * 4);
  float* z2   = alloc(4 * 1024 * 4);
  float* sqb  = alloc(1024 * 4);

  float *pA = bufA, *pB = bufB, *pC = bufC;

  auto CONV = [&](const float* in, float* out, const PW& c, int Ci, int Co,
                  int Hi, int Wi, int Ho, int Wo, int ks, int st, int pd) {
    dim3 g((Ho * Wo + CBN - 1) / CBN, (Co + CBM - 1) / CBM);
    conv_wmma_kernel<<<g, 256, 0, stream>>>(in, c.w, c.b, out,
                                            Ci, Co, Hi, Wi, Ho, Wo, ks, st, pd);
  };
  auto GN = [&](const float* in, float* out, const PGN& p, int C, int HW, int silu) {
    gn_kernel<<<32, 256, 0, stream>>>(in, out, p.g, p.b, HW, C / 32, silu);
  };
  auto ADD = [&](const float* a, const float* b, float* o, int n) {
    add_kernel<<<(n + 255) / 256, 256, 0, stream>>>(a, b, o, n);
  };
  auto RES = [&](const PRes& r, int Ci, int Co, int H, int W) {
    int HW = H * W;
    GN(pA, pB, r.n1, Ci, HW, 1);
    CONV(pB, pC, r.c1, Ci, Co, H, W, H, W, 3, 1, 1);
    GN(pC, pB, r.n2, Co, HW, 1);
    CONV(pB, pC, r.c2, Co, Co, H, W, H, W, 3, 1, 1);
    if (r.has_nin) {
      CONV(pA, pB, r.nin, Ci, Co, H, W, H, W, 1, 1, 0);
      ADD(pC, pB, pA, Co * HW);
    } else {
      ADD(pC, pA, pA, Co * HW);
    }
  };
  auto ATTN = [&](const PAttn& a) {
    const int C = 512, HW = 1024;
    GN(pA, pB, a.norm, C, HW, 0);
    CONV(pB, qb, a.q, C, C, 32, 32, 32, 32, 1, 1, 0);
    CONV(pB, kb, a.k, C, C, 32, 32, 32, 32, 1, 1, 0);
    CONV(pB, vb, a.v, C, C, 32, 32, 32, 32, 1, 1, 0);
    dim3 g1((1024 + CBN - 1) / CBN, (1024 + CBM - 1) / CBM);
    gemm_wmma_kernel<<<g1, 256, 0, stream>>>(qb, kb, attw, 1024, 1024, 512,
                                             1, 0, 0.044194173824159f); // 512^-0.5
    softmax_rows_kernel<<<1024, 256, 0, stream>>>(attw, 1024);
    dim3 g2((1024 + CBN - 1) / CBN, (512 + CBM - 1) / CBM);
    gemm_wmma_kernel<<<g2, 256, 0, stream>>>(vb, attw, pB, 512, 1024, 1024,
                                             0, 1, 1.0f);
    CONV(pB, pC, a.proj, C, C, 32, 32, 32, 32, 1, 1, 0);
    ADD(pA, pC, pA, C * HW);
  };

  // =================== encoder ===================
  CONV(x, pA, e_conv_in, 3, 128, 256, 256, 256, 256, 3, 1, 1);
  {
    const int chans[4] = {128, 256, 512, 512};
    int H = 256, Ci = 128;
    for (int i = 0; i < 4; ++i) {
      int Co = chans[i];
      RES(down[i][0], Ci, Co, H, H); Ci = Co;
      RES(down[i][1], Co, Co, H, H);
      if (i != 3) {
        int Ho = H / 2;
        // torch pad (0,1,0,1): pad end only -> padBeg=0, OOB reads are zero
        CONV(pA, pB, dsmp[i], Co, Co, H, H, Ho, Ho, 3, 2, 0);
        float* t = pA; pA = pB; pB = t;
        H = Ho;
      }
    }
  }
  RES(e_mb1, 512, 512, 32, 32);
  ATTN(e_attn);
  RES(e_mb2, 512, 512, 32, 32);
  GN(pA, pB, e_norm_out, 512, 1024, 1);
  CONV(pB, z0, e_conv_out, 512, 4, 32, 32, 32, 32, 3, 1, 1);

  // =================== quantize ===================
  CONV(z0, z1, quant, 4, 4, 32, 32, 32, 32, 1, 1, 0);
  quantize_kernel<<<(1024 + 127) / 128, 128, 0, stream>>>(z1, codebook, zqv, sqb,
                                                          1024, 16384);
  // diff = (1+BETA) * mean((zq-z)^2) over 4*32*32 elements
  diff_kernel<<<1, 256, 0, stream>>>(sqb, (float*)d_out + 3 * 256 * 256,
                                     1024, 1.25f / 4096.0f);
  CONV(zqv, z2, post_quant, 4, 4, 32, 32, 32, 32, 1, 1, 0);

  // =================== decoder ===================
  CONV(z2, pA, d_conv_in, 4, 512, 32, 32, 32, 32, 3, 1, 1);
  RES(d_mb1, 512, 512, 32, 32);
  ATTN(d_attn);
  RES(d_mb2, 512, 512, 32, 32);
  {
    const int chans[4] = {128, 256, 512, 512};
    int H = 32, Ci = 512;
    for (int i = 3; i >= 0; --i) {
      int Co = chans[i];
      RES(up[i][0], Ci, Co, H, H); Ci = Co;
      RES(up[i][1], Co, Co, H, H);
      RES(up[i][2], Co, Co, H, H);
      if (i != 0) {
        int n = Co * 4 * H * H;
        upsample_kernel<<<(n + 255) / 256, 256, 0, stream>>>(pA, pB, Co, H, H);
        CONV(pB, pC, upsmp[i], Co, Co, 2 * H, 2 * H, 2 * H, 2 * H, 3, 1, 1);
        float* t = pA; pA = pC; pC = t;
        H *= 2;
      }
    }
  }
  GN(pA, pB, d_norm_out, 128, 256 * 256, 1);
  CONV(pB, (float*)d_out, d_conv_out, 128, 3, 256, 256, 256, 256, 3, 1, 1);
}
